// FullyQuantumRNN_25013889532505
// MI455X (gfx1250) — matrix operand order repaired and checked by reading, not compiled, not run on hardware
//
#include <hip/hip_runtime.h>

typedef __attribute__((ext_vector_type(16))) _Float16 v16h;
typedef __attribute__((ext_vector_type(8)))  float    v8f;

struct cf { float x, y; };

__device__ __forceinline__ void apply2(cf& a, cf& b,
                                       const cf m00, const cf m01,
                                       const cf m10, const cf m11) {
  cf na{ m00.x*a.x - m00.y*a.y + m01.x*b.x - m01.y*b.y,
         m00.x*a.y + m00.y*a.x + m01.x*b.y + m01.y*b.x };
  cf nb{ m10.x*a.x - m10.y*a.y + m11.x*b.x - m11.y*b.y,
         m10.x*a.y + m10.y*a.x + m11.x*b.y + m11.y*b.x };
  a = na; b = nb;
}

__device__ __forceinline__ void rot_mat(float phi, float th, float om,
                                        cf& m00, cf& m01, cf& m10, cf& m11) {
  float ct, st; __sincosf(0.5f*th, &st, &ct);
  float ap = 0.5f*(phi+om), am = 0.5f*(phi-om);
  float cap, sap, cam, sam;
  __sincosf(ap, &sap, &cap); __sincosf(am, &sam, &cam);
  m00 = cf{  cap*ct, -sap*ct };   // exp(-i(phi+om)/2) * c
  m01 = cf{ -cam*st, -sam*st };   // -exp(+i(phi-om)/2) * s
  m10 = cf{  cam*st, -sam*st };   // exp(-i(phi-om)/2) * s
  m11 = cf{  cap*ct,  sap*ct };   // exp(+i(phi+om)/2) * c
}

// One-time: build A0,A1 = Re(U^H D_j U) (32x32, f16) and readout B0 (4x4, f32).
// 32 lanes; lane c owns column c of U (state-index i: wire w <-> bit (4-w)).
__global__ void qrnn_setup(const float* __restrict__ w_rec,
                           const float* __restrict__ w_out,
                           _Float16* __restrict__ a0,
                           _Float16* __restrict__ a1,
                           float* __restrict__ b0) {
  __shared__ cf U[32][33];                 // padded: lanes spread across banks
  const int c = threadIdx.x;
  #pragma unroll
  for (int i = 0; i < 32; ++i) U[c][i] = cf{ (i == c) ? 1.f : 0.f, 0.f };

  #pragma unroll
  for (int l = 0; l < 2; ++l) {
    #pragma unroll
    for (int w = 0; w < 5; ++w) {
      cf m00, m01, m10, m11;
      rot_mat(w_rec[(l*5+w)*3+0], w_rec[(l*5+w)*3+1], w_rec[(l*5+w)*3+2],
              m00, m01, m10, m11);
      const int bit = 1 << (4 - w);
      #pragma unroll
      for (int i0 = 0; i0 < 32; ++i0)
        if (!(i0 & bit)) apply2(U[c][i0], U[c][i0 | bit], m00, m01, m10, m11);
    }
    const int r = (l % 4) + 1;             // r = l % (n-1) + 1, n=5
    #pragma unroll
    for (int w = 0; w < 5; ++w) {
      const int tg = (w + r) % 5;
      const int cbit = 1 << (4 - w), tbit = 1 << (4 - tg);
      cf tmp[32];
      #pragma unroll
      for (int i = 0; i < 32; ++i) tmp[i] = U[c][(i & cbit) ? (i ^ tbit) : i];
      #pragma unroll
      for (int i = 0; i < 32; ++i) U[c][i] = tmp[i];
    }
  }
  __syncthreads();
  // A_j[R][c] = Re( sum_i conj(U[i][R]) z_j(i) U[i][c] );  wire3->bit1, wire4->bit0
  for (int R = 0; R < 32; ++R) {
    float acc0 = 0.f, acc1 = 0.f;
    #pragma unroll
    for (int i = 0; i < 32; ++i) {
      float re = U[R][i].x * U[c][i].x + U[R][i].y * U[c][i].y;
      acc0 += (i & 2) ? -re : re;
      acc1 += (i & 1) ? -re : re;
    }
    a0[R*32 + c] = (_Float16)acc0;
    a1[R*32 + c] = (_Float16)acc1;
  }
  // Readout observable (2 qubits): B0 = Re(V^H diag(z_w0) V), z(i)=(i&2)?-1:+1
  if (c == 0) {
    cf V[4][4];                            // V[col][i]
    #pragma unroll
    for (int cc = 0; cc < 4; ++cc)
      #pragma unroll
      for (int i = 0; i < 4; ++i) V[cc][i] = cf{ (i == cc) ? 1.f : 0.f, 0.f };
    #pragma unroll
    for (int l = 0; l < 2; ++l) {
      #pragma unroll
      for (int w = 0; w < 2; ++w) {
        cf m00, m01, m10, m11;
        rot_mat(w_out[(l*2+w)*3+0], w_out[(l*2+w)*3+1], w_out[(l*2+w)*3+2],
                m00, m01, m10, m11);
        const int bit = 1 << (1 - w);
        #pragma unroll
        for (int cc = 0; cc < 4; ++cc)
          #pragma unroll
          for (int i0 = 0; i0 < 4; ++i0)
            if (!(i0 & bit)) apply2(V[cc][i0], V[cc][i0 | bit], m00, m01, m10, m11);
      }
      #pragma unroll
      for (int w = 0; w < 2; ++w) {        // r == 1 always for n=2
        const int tg = (w + 1) % 2;
        const int cbit = 1 << (1 - w), tbit = 1 << (1 - tg);
        #pragma unroll
        for (int cc = 0; cc < 4; ++cc) {
          cf tmp[4];
          #pragma unroll
          for (int i = 0; i < 4; ++i) tmp[i] = V[cc][(i & cbit) ? (i ^ tbit) : i];
          #pragma unroll
          for (int i = 0; i < 4; ++i) V[cc][i] = tmp[i];
        }
      }
    }
    #pragma unroll
    for (int R = 0; R < 4; ++R)
      #pragma unroll
      for (int C = 0; C < 4; ++C) {
        float acc = 0.f;
        #pragma unroll
        for (int i = 0; i < 4; ++i) {
          float re = V[R][i].x * V[C][i].x + V[R][i].y * V[C][i].y;
          acc += (i & 2) ? -re : re;
        }
        b0[R*4 + C] = acc;
      }
  }
}

// Main recurrence. Wave = 16 batch columns; lane = (g = lane>>4, col = lane&15).
// Per step: 4x v_wmma_f32_16x16x32_f16 compute P_j = A_j * S (32x16), then
// h_j[b] = sum_i S[i,b] * P_j[i,b] via in-lane dot + shfl_xor across M-halves.
__global__ void __launch_bounds__(256) qrnn_main(const float* __restrict__ x,
                                                 const _Float16* __restrict__ a0,
                                                 const _Float16* __restrict__ a1,
                                                 const float* __restrict__ b0,
                                                 float* __restrict__ out,
                                                 int B, int T) {
  const int lane = threadIdx.x & 31;
  const int wv   = threadIdx.x >> 5;
  const int g    = lane >> 4;
  const int col  = lane & 15;
  const int b    = blockIdx.x * 128 + wv * 16 + col;
  const int bl   = (b < B) ? b : (B - 1);

  // A-operand fragments: lane row M=col,
  // half j=2v+p holds K = 16g + j.
  v16h fA0lo = {}, fA0hi = {}, fA1lo = {}, fA1hi = {};
  #pragma unroll
  for (int v = 0; v < 8; ++v)
    #pragma unroll
    for (int p = 0; p < 2; ++p) {
      const int K = 16 * g + 2 * v + p;
      const int j = 2 * v + p;
      fA0lo[j] = a0[ col       * 32 + K];
      fA0hi[j] = a0[(col + 16) * 32 + K];
      fA1lo[j] = a1[ col       * 32 + K];
      fA1hi[j] = a1[(col + 16) * 32 + K];
    }

  float h0 = 0.f, h1 = 0.f;
  for (int t = 0; t < T; ++t) {
    const float* xp = x + ((size_t)t * B + bl) * 3;
    float cw[5], sw[5];
    __sincosf(0.5f * xp[0], &sw[0], &cw[0]);
    __sincosf(0.5f * xp[1], &sw[1], &cw[1]);
    __sincosf(0.5f * xp[2], &sw[2], &cw[2]);
    __sincosf(0.5f * h0,    &sw[3], &cw[3]);
    __sincosf(0.5f * h1,    &sw[4], &cw[4]);

    // B-operand: s[k = 16g + j]; bits of k: 4->wire0, 3..0 -> wires1..4
    v16h bf = {};
    const float f0 = g ? sw[0] : cw[0];
    #pragma unroll
    for (int j = 0; j < 16; ++j) {
      float p = f0 * ((j & 8) ? sw[1] : cw[1])
                   * ((j & 4) ? sw[2] : cw[2])
                   * ((j & 2) ? sw[3] : cw[3])
                   * ((j & 1) ? sw[4] : cw[4]);
      bf[j] = (_Float16)p;
    }
    // C-layout copy of s: row i = 16r + 8g + v
    float sc0[8], sc1[8];
    const float f1 = g ? sw[1] : cw[1];
    #pragma unroll
    for (int v = 0; v < 8; ++v) {
      float base = f1 * ((v & 4) ? sw[2] : cw[2])
                      * ((v & 2) ? sw[3] : cw[3])
                      * ((v & 1) ? sw[4] : cw[4]);
      sc0[v] = cw[0] * base;
      sc1[v] = sw[0] * base;
    }

    const v8f z = {};
    v8f P0lo = __builtin_amdgcn_wmma_f32_16x16x32_f16(false, fA0lo, false, bf, (short)0, z, false, false);
    v8f P0hi = __builtin_amdgcn_wmma_f32_16x16x32_f16(false, fA0hi, false, bf, (short)0, z, false, false);
    v8f P1lo = __builtin_amdgcn_wmma_f32_16x16x32_f16(false, fA1lo, false, bf, (short)0, z, false, false);
    v8f P1hi = __builtin_amdgcn_wmma_f32_16x16x32_f16(false, fA1hi, false, bf, (short)0, z, false, false);

    float p0 = 0.f, p1 = 0.f;
    #pragma unroll
    for (int v = 0; v < 8; ++v) {
      p0 += sc0[v] * P0lo[v] + sc1[v] * P0hi[v];
      p1 += sc0[v] * P1lo[v] + sc1[v] * P1hi[v];
    }
    h0 = p0 + __shfl_xor(p0, 16);
    h1 = p1 + __shfl_xor(p1, 16);
  }

  // Readout: out = t^T B0 t with t = kron((c0,s0),(c1,s1))
  float c0, s0, c1, s1;
  __sincosf(0.5f * h0, &s0, &c0);
  __sincosf(0.5f * h1, &s1, &c1);
  const float tv[4] = { c0*c1, c0*s1, s0*c1, s0*s1 };
  float o = 0.f;
  #pragma unroll
  for (int i = 0; i < 4; ++i)
    #pragma unroll
    for (int k = 0; k < 4; ++k)
      o += tv[i] * b0[i*4 + k] * tv[k];
  if (g == 0 && b < B) out[b] = o;
}

extern "C" void kernel_launch(void* const* d_in, const int* in_sizes, int n_in,
                              void* d_out, int out_size, void* d_ws, size_t ws_size,
                              hipStream_t stream) {
  const float* x_seq = (const float*)d_in[0];   // (T, B, 3) f32
  const float* w_rec = (const float*)d_in[1];   // (2, 5, 3) f32
  const float* w_out = (const float*)d_in[2];   // (2, 2, 3) f32
  float* out = (float*)d_out;                   // (B, 1) f32

  const int B = out_size;
  const int T = in_sizes[0] / (B * 3);

  _Float16* a0 = (_Float16*)d_ws;               // 32x32 f16
  _Float16* a1 = a0 + 32 * 32;                  // 32x32 f16
  float*    b0 = (float*)((char*)d_ws + 4096);  // 4x4 f32

  qrnn_setup<<<1, 32, 0, stream>>>(w_rec, w_out, a0, a1, b0);
  const int blocks = (B + 127) / 128;
  qrnn_main<<<blocks, 256, 0, stream>>>(x_seq, a0, a1, b0, out, B, T);
}
